// HybridGPT_55301998903671
// MI455X (gfx1250) — matrix operand order repaired and checked by reading, not compiled, not run on hardware
//
#include <hip/hip_runtime.h>

#define DIMD   384
#define STATEN 384
#define SELHD  128
#define NEXPE  4
#define BB     16
#define SSEQ   2048
#define BSTOK  (BB*SSEQ)      // 32768 tokens
#define DFF    (DIMD*3)       // 1152
#define SO4    (4*STATEN)     // 1536

typedef __bf16 v16bf __attribute__((ext_vector_type(16)));
typedef __bf16 v8bf  __attribute__((ext_vector_type(8)));
typedef float  v8f   __attribute__((ext_vector_type(8)));

// ---------------------------------------------------------------- helpers
__device__ __forceinline__ unsigned short f2bf(float f) {
  unsigned int u = __float_as_uint(f);
  return (unsigned short)((u + 0x7FFFu + ((u >> 16) & 1u)) >> 16);  // RNE
}

__device__ __forceinline__ float sigmoidf_(float v) {
  return 1.0f / (1.0f + __expf(-v));
}

__device__ __forceinline__ int murmur_route(unsigned int x) {
  x ^= x >> 16; x *= 2246822507u;
  x ^= x >> 13; x *= 3266489909u;
  x ^= x >> 16;
  return (int)(x % (unsigned)NEXPE);
}

__device__ __forceinline__ v8f wmma_bf16(v16bf a, v16bf b, v8f c) {
  return __builtin_amdgcn_wmma_f32_16x16x32_bf16(false, a, false, b,
                                                 (short)0, c, false, false);
}

// Direct global -> LDS async copy (8 bytes / lane), tracked by ASYNCcnt.
// LDS offset = low 32 bits of the flat shared pointer (wave-relative, ISA 10.2).
__device__ __forceinline__ void async_copy_b64(unsigned lds_off, const void* gaddr) {
  asm volatile("global_load_async_to_lds_b64 %0, %1, off"
               :
               : "v"(lds_off), "v"(gaddr)
               : "memory");
}
__device__ __forceinline__ void wait_async_all() {
  asm volatile("s_wait_asynccnt 0x0" ::: "memory");
}

// A-fragment (16x32 bf16) from an LDS tile stored row-major [16][32].
// ISA layout: lanes 0-15 hold row=lane, K 0-7 & 16-23; lanes 16-31 hold K 8-15 & 24-31.
__device__ __forceinline__ v16bf afrag_from_lds(const unsigned short* As, int lane) {
  int r  = lane & 15;
  int kb = (lane >> 4) << 3;
  v8bf L = *(const v8bf*)(As + r * 32 + kb);
  v8bf H = *(const v8bf*)(As + r * 32 + 16 + kb);
  v16bf a;
#pragma unroll
  for (int i = 0; i < 8; ++i) { a[i] = L[i]; a[i + 8] = H[i]; }
  return a;
}

__device__ __forceinline__ float sel4(const v8f* acc, int r, int v) {
  float x0 = acc[0][v], x1 = acc[1][v], x2 = acc[2][v], x3 = acc[3][v];
  float lo = (r & 1) ? x1 : x0;
  float hi = (r & 1) ? x3 : x2;
  return (r & 2) ? hi : lo;
}

// Shared GEMM core: block = 128 threads (4 waves); block owns 16 rows (m0),
// each wave one 16-col tile (nt); E experts accumulated against one A-frag.
// Double-buffered LDS A tile staged via async global->LDS; next-step B frags
// prefetched into registers so the async copy + B loads overlap the WMMAs.
template <int E>
__device__ __forceinline__ void run_gemm(const unsigned short* __restrict__ Ab, int K,
                                         const unsigned short* __restrict__ Wp,
                                         size_t wstride, int m0, int nt,
                                         unsigned short* As /*[2*512]*/, v8f* acc) {
  const int lane   = threadIdx.x & 31;
  const int ktiles = K >> 5;
  const int t      = threadIdx.x;
  const unsigned short* arow = Ab + (size_t)(m0 + (t >> 3)) * K + ((t & 7) << 2);
  const unsigned lds0 = (unsigned)(size_t)As + (unsigned)t * 8u;

#pragma unroll
  for (int e = 0; e < E; ++e)
#pragma unroll
    for (int i = 0; i < 8; ++i) acc[e][i] = 0.0f;

  // prologue: stage tile 0, preload B frags for kt=0
  async_copy_b64(lds0, arow);
  v16bf bf[E];
#pragma unroll
  for (int e = 0; e < E; ++e)
    bf[e] = *(const v16bf*)(Wp + (size_t)e * wstride +
                            ((size_t)(nt * ktiles) << 9) + (lane << 4));
  wait_async_all();
  __syncthreads();

  for (int kt = 0; kt < ktiles; ++kt) {
    const int cur = kt & 1;
    const bool more = (kt + 1) < ktiles;
    if (more)   // async stage of next A tile runs under this step's WMMAs
      async_copy_b64(lds0 + (unsigned)((cur ^ 1) * 1024), arow + ((kt + 1) << 5));

    v16bf af = afrag_from_lds(As + cur * 512, lane);
    v16bf bnext[E];
    if (more) {
#pragma unroll
      for (int e = 0; e < E; ++e)
        bnext[e] = *(const v16bf*)(Wp + (size_t)e * wstride +
                                   ((size_t)(nt * ktiles + kt + 1) << 9) + (lane << 4));
    }
#pragma unroll
    for (int e = 0; e < E; ++e) acc[e] = wmma_bf16(af, bf[e], acc[e]);
    if (more) {
#pragma unroll
      for (int e = 0; e < E; ++e) bf[e] = bnext[e];
      wait_async_all();
    }
    __syncthreads();
  }
}

// ---------------------------------------------------------------- kernels

// Repack f32 weight [E][K][N] into WMMA-native bf16 tiles:
// tile id = nt*ktiles + kt; lane l holds B[kt*32 + (i>>3)*16 + (l>>4)*8 + (i&7)][nt*16 + (l&15)]
__global__ void k_repack(const float* __restrict__ W, unsigned short* __restrict__ Wp,
                         int K, int N, int nexp) {
  size_t per   = (size_t)K * N;
  size_t total = (size_t)nexp * per;
  int ktiles   = K >> 5;
  for (size_t o = (size_t)blockIdx.x * blockDim.x + threadIdx.x; o < total;
       o += (size_t)gridDim.x * blockDim.x) {
    size_t e = o / per;
    size_t i = o % per;
    int tile = (int)(i >> 9);
    int lane = (int)((i >> 4) & 31);
    int el   = (int)(i & 15);
    int kt   = tile % ktiles;
    int nt   = tile / ktiles;
    int row  = (kt << 5) + ((el >> 3) << 4) + ((lane >> 4) << 3) + (el & 7);
    int col  = (nt << 4) + (lane & 15);
    Wp[o] = f2bf(W[e * per + (size_t)row * N + col]);
  }
}

// RMSNorm (one wave per token) -> bf16, optionally also computes murmur routes.
__global__ void k_norm_route(const float* __restrict__ x, const int* __restrict__ ids,
                             const float* __restrict__ w, unsigned short* __restrict__ xn,
                             int* __restrict__ routes, int do_routes) {
  int lane = threadIdx.x & 31;
  int tok  = blockIdx.x * (blockDim.x >> 5) + (threadIdx.x >> 5);
  if (tok >= BSTOK) return;
  const float* xr = x + (size_t)tok * DIMD;
  float ss = 0.0f;
  for (int i = lane; i < DIMD; i += 32) { float v = xr[i]; ss += v * v; }
#pragma unroll
  for (int off = 16; off > 0; off >>= 1) ss += __shfl_xor(ss, off, 32);
  float scale = rsqrtf(ss * (1.0f / DIMD) + 1e-6f);
  unsigned short* xo = xn + (size_t)tok * DIMD;
  for (int i = lane; i < DIMD; i += 32) xo[i] = f2bf(xr[i] * scale * w[i]);
  if (do_routes && lane == 0) routes[tok] = murmur_route((unsigned int)ids[tok]);
}

// u = xn @ W_in[route]   (4 dense accs, per-row select)
__global__ void k_gemm_u(const unsigned short* __restrict__ xn,
                         const unsigned short* __restrict__ Wp,
                         const int* __restrict__ routes, float* __restrict__ u) {
  __shared__ __attribute__((aligned(16))) unsigned short As[2 * 512];
  int lane = threadIdx.x & 31, wave = threadIdx.x >> 5;
  int m0 = blockIdx.x << 4, nt = (blockIdx.y << 2) + wave;
  v8f acc[NEXPE];
  run_gemm<NEXPE>(xn, DIMD, Wp, (size_t)DIMD * STATEN, m0, nt, As, acc);
  int cl = lane & 15;
#pragma unroll
  for (int v = 0; v < 8; ++v) {
    int row = m0 + ((lane >> 4) << 3) + v;
    int r = routes[row];
    u[(size_t)row * STATEN + (nt << 4) + cl] = sel4(acc, r, v);
  }
}

// sh[e] = silu(xn @ W_sel_in[e])   (all 4 experts kept, bf16)
__global__ void k_gemm_selin(const unsigned short* __restrict__ xn,
                             const unsigned short* __restrict__ Wp,
                             unsigned short* __restrict__ sh) {
  __shared__ __attribute__((aligned(16))) unsigned short As[2 * 512];
  int lane = threadIdx.x & 31, wave = threadIdx.x >> 5;
  int m0 = blockIdx.x << 4, nt = (blockIdx.y << 2) + wave;
  v8f acc[NEXPE];
  run_gemm<NEXPE>(xn, DIMD, Wp, (size_t)DIMD * SELHD, m0, nt, As, acc);
  int cl = lane & 15;
#pragma unroll
  for (int v = 0; v < 8; ++v) {
    int row = m0 + ((lane >> 4) << 3) + v;
    size_t o = (size_t)row * SELHD + (nt << 4) + cl;
#pragma unroll
    for (int e = 0; e < NEXPE; ++e) {
      float s = acc[e][v];
      sh[(size_t)e * BSTOK * SELHD + o] = f2bf(s * sigmoidf_(s));
    }
  }
}

// so = sh[e] @ W_sel_out[e], per-row expert select, fused gate epilogue:
// cols [0,384):a=sig  [384,768):bu=tanh*u  [768,1152):c=tanh  [1152,1536):du=sig*d*u
__global__ void k_gemm_selout(const unsigned short* __restrict__ sh,
                              const unsigned short* __restrict__ Wp,
                              const int* __restrict__ routes,
                              const float* __restrict__ u,
                              const float* __restrict__ d_param,
                              float* __restrict__ Ab, float* __restrict__ BUb,
                              float* __restrict__ Cb, float* __restrict__ DUb) {
  __shared__ __attribute__((aligned(16))) unsigned short As[2][NEXPE][512];
  int lane = threadIdx.x & 31, wave = threadIdx.x >> 5;
  int m0 = blockIdx.x << 4, nt = (blockIdx.y << 2) + wave;
  const int ktiles = SELHD >> 5;   // 4
  const int t = threadIdx.x;
  const size_t arow = (size_t)(m0 + (t >> 3)) * SELHD + ((t & 7) << 2);
  const unsigned lds0 = (unsigned)(size_t)(&As[0][0][0]) + (unsigned)t * 8u;

  v8f acc[NEXPE];
#pragma unroll
  for (int e = 0; e < NEXPE; ++e)
#pragma unroll
    for (int i = 0; i < 8; ++i) acc[e][i] = 0.0f;

  // prologue: stage all 4 expert A tiles for kt=0, preload B frags
#pragma unroll
  for (int e = 0; e < NEXPE; ++e)
    async_copy_b64(lds0 + (unsigned)(e * 1024),
                   sh + (size_t)e * BSTOK * SELHD + arow);
  v16bf bf[NEXPE];
#pragma unroll
  for (int e = 0; e < NEXPE; ++e)
    bf[e] = *(const v16bf*)(Wp + (size_t)e * SELHD * SO4 +
                            ((size_t)(nt * ktiles) << 9) + (lane << 4));
  wait_async_all();
  __syncthreads();

  for (int kt = 0; kt < ktiles; ++kt) {
    const int cur = kt & 1;
    const bool more = (kt + 1) < ktiles;
    if (more) {
#pragma unroll
      for (int e = 0; e < NEXPE; ++e)
        async_copy_b64(lds0 + (unsigned)((cur ^ 1) * 4096 + e * 1024),
                       sh + (size_t)e * BSTOK * SELHD + arow + ((kt + 1) << 5));
    }
    v16bf bnext[NEXPE];
    if (more) {
#pragma unroll
      for (int e = 0; e < NEXPE; ++e)
        bnext[e] = *(const v16bf*)(Wp + (size_t)e * SELHD * SO4 +
                                   ((size_t)(nt * ktiles + kt + 1) << 9) + (lane << 4));
    }
#pragma unroll
    for (int e = 0; e < NEXPE; ++e) {
      v16bf af = afrag_from_lds(&As[cur][e][0], lane);
      acc[e] = wmma_bf16(af, bf[e], acc[e]);
    }
    if (more) {
#pragma unroll
      for (int e = 0; e < NEXPE; ++e) bf[e] = bnext[e];
      wait_async_all();
    }
    __syncthreads();
  }

  int cl = lane & 15;
  int n  = (nt << 4) + cl;     // 0..1535
  int g  = n / STATEN;         // gate id (uniform per wave)
  int j  = n % STATEN;
#pragma unroll
  for (int v = 0; v < 8; ++v) {
    int row = m0 + ((lane >> 4) << 3) + v;
    int r = routes[row];
    float val = sel4(acc, r, v);
    size_t o = (size_t)row * STATEN + j;
    if (g == 0)      Ab[o]  = sigmoidf_(val);
    else if (g == 1) BUb[o] = tanhf(val) * u[o];
    else if (g == 2) Cb[o]  = tanhf(val);
    else             DUb[o] = sigmoidf_(val) * d_param[r * STATEN + j] * u[o];
  }
}

// Selective scan: thread = (b,n); 4 per-expert states in registers, y -> bf16.
__global__ void k_scan(const int* __restrict__ routes, const float* __restrict__ Ab,
                       const float* __restrict__ BUb, const float* __restrict__ Cb,
                       const float* __restrict__ DUb, unsigned short* __restrict__ y) {
  int g = blockIdx.x * blockDim.x + threadIdx.x;
  if (g >= BB * STATEN) return;
  int b = g / STATEN, n = g % STATEN;
  float h0 = 0.f, h1 = 0.f, h2 = 0.f, h3 = 0.f;
  const size_t base = (size_t)b * SSEQ * STATEN + n;
  const int* rp = routes + b * SSEQ;
  for (int t = 0; t < SSEQ; ++t) {
    size_t idx = base + (size_t)t * STATEN;
    if (t + 16 < SSEQ) {                       // global_prefetch_b8 lookahead
      size_t p = base + (size_t)(t + 16) * STATEN;
      __builtin_prefetch(Ab + p, 0, 0);
      __builtin_prefetch(BUb + p, 0, 0);
      __builtin_prefetch(Cb + p, 0, 0);
      __builtin_prefetch(DUb + p, 0, 0);
    }
    int r = rp[t];
    float a = Ab[idx], bu = BUb[idx];
    float hp = (r == 0) ? h0 : (r == 1) ? h1 : (r == 2) ? h2 : h3;
    float hn = a * hp + bu;
    if (r == 0) h0 = hn; else if (r == 1) h1 = hn;
    else if (r == 2) h2 = hn; else h3 = hn;
    y[idx] = f2bf(Cb[idx] * hn + DUb[idx]);
  }
}

// x1 = x + ssm_scale * (y @ W_out[route])
__global__ void k_gemm_out(const unsigned short* __restrict__ y,
                           const unsigned short* __restrict__ Wp,
                           const int* __restrict__ routes, const float* __restrict__ x,
                           const float* __restrict__ ssm_sc, float* __restrict__ x1) {
  __shared__ __attribute__((aligned(16))) unsigned short As[2 * 512];
  int lane = threadIdx.x & 31, wave = threadIdx.x >> 5;
  int m0 = blockIdx.x << 4, nt = (blockIdx.y << 2) + wave;
  v8f acc[NEXPE];
  run_gemm<NEXPE>(y, STATEN, Wp, (size_t)STATEN * DIMD, m0, nt, As, acc);
  int cl = lane & 15, n = (nt << 4) + cl;
#pragma unroll
  for (int v = 0; v < 8; ++v) {
    int row = m0 + ((lane >> 4) << 3) + v;
    int r = routes[row];
    size_t o = (size_t)row * DIMD + n;
    x1[o] = x[o] + ssm_sc[n] * sel4(acc, r, v);
  }
}

// hid = relu(mn @ W_fc)^2 -> bf16
__global__ void k_gemm_fc(const unsigned short* __restrict__ mn,
                          const unsigned short* __restrict__ Wp,
                          unsigned short* __restrict__ hid) {
  __shared__ __attribute__((aligned(16))) unsigned short As[2 * 512];
  int lane = threadIdx.x & 31, wave = threadIdx.x >> 5;
  int m0 = blockIdx.x << 4, nt = (blockIdx.y << 2) + wave;
  v8f acc[1];
  run_gemm<1>(mn, DIMD, Wp, 0, m0, nt, As, acc);
  int cl = lane & 15, n = (nt << 4) + cl;
#pragma unroll
  for (int v = 0; v < 8; ++v) {
    int row = m0 + ((lane >> 4) << 3) + v;
    float t = fmaxf(acc[0][v], 0.0f);
    hid[(size_t)row * DFF + n] = f2bf(t * t);
  }
}

// out = x1 + mlp_scale * (hid @ W_proj)
__global__ void k_gemm_proj(const unsigned short* __restrict__ hid,
                            const unsigned short* __restrict__ Wp,
                            const float* __restrict__ x1,
                            const float* __restrict__ mlp_sc, float* __restrict__ out) {
  __shared__ __attribute__((aligned(16))) unsigned short As[2 * 512];
  int lane = threadIdx.x & 31, wave = threadIdx.x >> 5;
  int m0 = blockIdx.x << 4, nt = (blockIdx.y << 2) + wave;
  v8f acc[1];
  run_gemm<1>(hid, DFF, Wp, 0, m0, nt, As, acc);
  int cl = lane & 15, n = (nt << 4) + cl;
#pragma unroll
  for (int v = 0; v < 8; ++v) {
    int row = m0 + ((lane >> 4) << 3) + v;
    size_t o = (size_t)row * DIMD + n;
    out[o] = x1[o] + mlp_sc[n] * acc[0][v];
  }
}

// ---------------------------------------------------------------- host
extern "C" void kernel_launch(void* const* d_in, const int* in_sizes, int n_in,
                              void* d_out, int out_size, void* d_ws, size_t ws_size,
                              hipStream_t stream) {
  (void)in_sizes; (void)n_in; (void)out_size; (void)ws_size;
  const float* x       = (const float*)d_in[0];
  const int*   ids     = (const int*)d_in[1];
  const float* W_in    = (const float*)d_in[2];
  const float* W_si    = (const float*)d_in[3];
  const float* W_so    = (const float*)d_in[4];
  const float* W_out   = (const float*)d_in[5];
  const float* d_param = (const float*)d_in[6];
  const float* ssm_nw  = (const float*)d_in[7];
  const float* mlp_nw  = (const float*)d_in[8];
  const float* ssm_sc  = (const float*)d_in[9];
  const float* mlp_sc  = (const float*)d_in[10];
  const float* W_fc    = (const float*)d_in[11];
  const float* W_pj    = (const float*)d_in[12];
  float* out = (float*)d_out;

  char* ws = (char*)d_ws;
  size_t off = 0;
  auto alloc = [&](size_t bytes) {
    size_t o = off;
    off = (off + bytes + 255) & ~(size_t)255;
    return ws + o;
  };
  int*            routes = (int*)alloc((size_t)BSTOK * 4);
  unsigned short* xn     = (unsigned short*)alloc((size_t)BSTOK * DIMD * 2);   // also mn
  float*          u      = (float*)alloc((size_t)BSTOK * STATEN * 4);
  unsigned short* sh     = (unsigned short*)alloc((size_t)NEXPE * BSTOK * SELHD * 2);
  float*          Ab     = (float*)alloc((size_t)BSTOK * STATEN * 4);
  float*          BUb    = (float*)alloc((size_t)BSTOK * STATEN * 4);
  float*          Cb     = (float*)alloc((size_t)BSTOK * STATEN * 4);
  float*          DUb    = (float*)alloc((size_t)BSTOK * STATEN * 4);
  unsigned short* ybf    = (unsigned short*)alloc((size_t)BSTOK * STATEN * 2);
  float*          x1     = (float*)alloc((size_t)BSTOK * DIMD * 4);
  unsigned short* Winp   = (unsigned short*)alloc((size_t)NEXPE * DIMD * STATEN * 2);
  unsigned short* Wsip   = (unsigned short*)alloc((size_t)NEXPE * DIMD * SELHD * 2);
  unsigned short* Wsop   = (unsigned short*)alloc((size_t)NEXPE * SELHD * SO4 * 2);
  unsigned short* Wop    = (unsigned short*)alloc((size_t)NEXPE * STATEN * DIMD * 2);
  unsigned short* Wfcp   = (unsigned short*)alloc((size_t)DIMD * DFF * 2);
  unsigned short* Wpjp   = (unsigned short*)alloc((size_t)DFF * DIMD * 2);
  // hid (BSTOK*DFF bf16 = 72 MB) aliases Ab+BUb (96 MB), dead after k_scan.
  unsigned short* hid = (unsigned short*)Ab;

  // 1) one-time weight repack into WMMA-native bf16 tile layout
  k_repack<<<256, 256, 0, stream>>>(W_in, Winp, DIMD, STATEN, NEXPE);
  k_repack<<<256, 256, 0, stream>>>(W_si, Wsip, DIMD, SELHD, NEXPE);
  k_repack<<<256, 256, 0, stream>>>(W_so, Wsop, SELHD, SO4, NEXPE);
  k_repack<<<256, 256, 0, stream>>>(W_out, Wop, STATEN, DIMD, NEXPE);
  k_repack<<<256, 256, 0, stream>>>(W_fc, Wfcp, DIMD, DFF, 1);
  k_repack<<<256, 256, 0, stream>>>(W_pj, Wpjp, DFF, DIMD, 1);

  // 2) routes + ssm RMSNorm -> bf16
  k_norm_route<<<BSTOK / 8, 256, 0, stream>>>(x, ids, ssm_nw, xn, routes, 1);

  // 3) expert projections (WMMA)
  k_gemm_u<<<dim3(BSTOK / 16, STATEN / 64), 128, 0, stream>>>(xn, Winp, routes, u);
  k_gemm_selin<<<dim3(BSTOK / 16, SELHD / 64), 128, 0, stream>>>(xn, Wsip, sh);
  k_gemm_selout<<<dim3(BSTOK / 16, SO4 / 64), 128, 0, stream>>>(
      sh, Wsop, routes, u, d_param, Ab, BUb, Cb, DUb);

  // 4) selective scan (register-resident expert states)
  k_scan<<<(BB * STATEN + 127) / 128, 128, 0, stream>>>(routes, Ab, BUb, Cb, DUb, ybf);

  // 5) output projection + residual
  k_gemm_out<<<dim3(BSTOK / 16, DIMD / 64), 128, 0, stream>>>(
      ybf, Wop, routes, x, ssm_sc, x1);

  // 6) MLP
  k_norm_route<<<BSTOK / 8, 256, 0, stream>>>(x1, nullptr, mlp_nw, xn, nullptr, 0);
  k_gemm_fc<<<dim3(BSTOK / 16, DFF / 64), 128, 0, stream>>>(xn, Wfcp, hid);
  k_gemm_proj<<<dim3(BSTOK / 16, DIMD / 64), 128, 0, stream>>>(hid, Wpjp, x1, mlp_sc, out);
}
